// SHDDelaysSNN_15831249453622
// MI455X (gfx1250) — compile-verified, matrix operand order checked by reading, not run on hardware
//
#include <hip/hip_runtime.h>
#include <hip/hip_bf16.h>

// ---------------- problem dimensions ----------------
#define Bsz   256
#define Tlen  250
#define Tpad  256
#define Cin   700
#define Cpad  704       // pad C to multiple of 32 for K-loop
#define H1    1024
#define H2    512
#define Osz   20

typedef __attribute__((ext_vector_type(16))) __bf16        v16bf;
typedef __attribute__((ext_vector_type(8)))  float         v8f;
typedef __attribute__((ext_vector_type(8)))  unsigned short v8us;

// f32 -> bf16 (round to nearest even)
static __device__ __forceinline__ unsigned short f2bf(float f) {
  unsigned u = __builtin_bit_cast(unsigned, f);
  unsigned r = u + 0x7FFFu + ((u >> 16) & 1u);
  return (unsigned short)(r >> 16);
}
static __device__ __forceinline__ float bf2f(unsigned short s) {
  unsigned u = ((unsigned)s) << 16;
  return __builtin_bit_cast(float, u);
}

// Load 16 bf16 for one lane of a WMMA A/B operand (two 16B chunks: k, k+16)
static __device__ __forceinline__ v16bf load_ab(const unsigned short* p) {
  union { v8us h[2]; v16bf v; } u;
  u.h[0] = *reinterpret_cast<const v8us*>(p);
  u.h[1] = *reinterpret_cast<const v8us*>(p + 16);
  return u.v;
}

#define V8ZERO {0.f,0.f,0.f,0.f,0.f,0.f,0.f,0.f}
#define WMMA_BF16(a, b, c) \
  __builtin_amdgcn_wmma_f32_16x16x32_bf16(false, (a), false, (b), (short)0, (c), false, false)

// ---------------- weight conversion ----------------
__global__ void k_f32_to_bf16(const float* __restrict__ src,
                              unsigned short* __restrict__ dst, int n) {
  int i = blockIdx.x * blockDim.x + threadIdx.x;
  if (i < n) dst[i] = f2bf(src[i]);
}

__global__ void k_convert_win(const float* __restrict__ W_in,
                              unsigned short* __restrict__ dst) {
  int i = blockIdx.x * blockDim.x + threadIdx.x;
  if (i >= H1 * Cpad) return;
  int h = i / Cpad, c = i % Cpad;
  dst[i] = (c < Cin) ? f2bf(W_in[h * Cin + c]) : (unsigned short)0;
}

// ---------------- per-channel delay shift (binary spikes -> exact bf16) ----
__global__ void k_delay(const float* __restrict__ x, const int* __restrict__ delays,
                        unsigned short* __restrict__ xd) {
  long i = (long)blockIdx.x * blockDim.x + threadIdx.x;
  if (i >= (long)Bsz * Tpad * Cpad) return;
  int c = (int)(i % Cpad);
  int t = (int)((i / Cpad) % Tpad);
  int b = (int)(i / ((long)Cpad * Tpad));
  unsigned short v = 0;
  if (c < Cin && t < Tlen) {
    int ts = t - delays[c];
    if (ts >= 0) v = f2bf(x[((long)b * Tlen + ts) * Cin + c]);
  }
  xd[i] = v;
}

// ---------------- input GEMM: I_in[b,t,h] = xd[b,t,:] . W_in[h,:] ----------
__global__ __launch_bounds__(256) void k_input_gemm(
    const unsigned short* __restrict__ xd,
    const unsigned short* __restrict__ Winb,
    float* __restrict__ Iin) {
  __shared__ __align__(16) unsigned short Ap[16 * Cpad];  // 22 KB panel
  const int tid = threadIdx.x, lane = tid & 31, wave = tid >> 5;
  const int b = blockIdx.x, t0 = blockIdx.y * 16;
  const int m16  = lane & 15;
  const int koff = (lane & 16) ? 8 : 0;
  const int moff = (lane & 16) ? 8 : 0;

  // cooperative 128-bit panel load (16 contiguous rows of xd)
  {
    const v8us* src = reinterpret_cast<const v8us*>(xd + ((long)b * Tpad + t0) * Cpad);
    v8us* dst = reinterpret_cast<v8us*>(Ap);
    for (int i = tid; i < 16 * Cpad / 8; i += 256) dst[i] = src[i];
  }
  __syncthreads();

  for (int tt = 0; tt < 8; ++tt) {            // 8 waves x 8 tiles = H1/16
    const int n0 = (wave * 8 + tt) * 16;
    v8f c = V8ZERO;
    for (int k0 = 0; k0 < Cpad; k0 += 32) {
      v16bf A  = load_ab(&Ap[m16 * Cpad + k0 + koff]);
      v16bf Bv = load_ab(&Winb[(n0 + m16) * Cpad + k0 + koff]);
      c = WMMA_BF16(A, Bv, c);
    }
    for (int r = 0; r < 8; ++r) {
      int m = r + moff, h = n0 + m16;
      Iin[((long)b * Tpad + (t0 + m)) * H1 + h] = c[r];
    }
  }
}

// ---------------- persistent recurrent scan ---------------------------------
// One WG per 16-batch tile; recurrence is batch-independent -> no grid sync.
__global__ __launch_bounds__(512) void k_recurrent(
    const float* __restrict__ Iin,
    const unsigned short* __restrict__ Wr1,
    const unsigned short* __restrict__ W2b,
    const unsigned short* __restrict__ Wr2,
    const float* __restrict__ Wout,
    const float* __restrict__ alpha1, const float* __restrict__ rho1,
    const float* __restrict__ beta_a1,
    const float* __restrict__ alpha2, const float* __restrict__ rho2,
    const float* __restrict__ beta_a2,
    const float* __restrict__ beta_out,
    float* __restrict__ v1, float* __restrict__ a1,
    float* __restrict__ v2, float* __restrict__ a2,
    float* __restrict__ out) {
  __shared__ __align__(16) unsigned short s1_sh[16 * H1];  // 32 KB bf16 spikes
  __shared__ __align__(16) unsigned short s2_sh[16 * H2];  // 16 KB
  __shared__ float vo_sh[16 * Osz];
  __shared__ float acc_sh[16 * Osz];

  const int tid  = threadIdx.x;
  const int lane = tid & 31;
  const int wave = tid >> 5;                  // 16 waves
  const int b0   = blockIdx.x * 16;
  const int m16  = lane & 15;
  const int koff = (lane & 16) ? 8 : 0;
  const int moff = (lane & 16) ? 8 : 0;

  // zero-init state (workspace is poisoned; we own these rows exclusively)
  for (int i = tid; i < 16 * H1; i += 512) {
    s1_sh[i] = 0;
    int bb = i / H1, hh = i % H1;
    v1[(b0 + bb) * H1 + hh] = 0.f;
    a1[(b0 + bb) * H1 + hh] = 0.f;
  }
  for (int i = tid; i < 16 * H2; i += 512) {
    s2_sh[i] = 0;
    int bb = i / H2, hh = i % H2;
    v2[(b0 + bb) * H2 + hh] = 0.f;
    a2[(b0 + bb) * H2 + hh] = 0.f;
  }
  for (int i = tid; i < 16 * Osz; i += 512) { vo_sh[i] = 0.f; acc_sh[i] = 0.f; }
  __syncthreads();

  for (int t = 0; t < Tlen; ++t) {
    // ---- layer 1: I1 = I_in[t] + s1 @ W_rec1.T   (4 N-tiles per wave)
    v8f acc1[4];
    for (int tt = 0; tt < 4; ++tt) {
      const int n0 = (wave * 4 + tt) * 16;
      v8f c = V8ZERO;
      for (int k0 = 0; k0 < H1; k0 += 32) {
        __builtin_prefetch((const void*)&Wr1[(n0 + m16) * H1 + k0 + 64], 0, 0);
        v16bf A  = load_ab(&s1_sh[m16 * H1 + k0 + koff]);
        v16bf Bv = load_ab(&Wr1[(n0 + m16) * H1 + k0 + koff]);
        c = WMMA_BF16(A, Bv, c);
      }
      acc1[tt] = c;
    }
    __syncthreads();

    // ---- layer 1 adLIF update + spike (soft reset), write new spikes to LDS
    for (int tt = 0; tt < 4; ++tt) {
      const int n0 = (wave * 4 + tt) * 16;
      const int h  = n0 + m16;
      const float al = alpha1[h], rh = rho1[h], be = beta_a1[h];
      v8f c = acc1[tt];
      for (int r = 0; r < 8; ++r) {
        const int m = r + moff, gb = b0 + m;
        float I1 = c[r] + Iin[((long)gb * Tpad + t) * H1 + h];
        float vv = v1[gb * H1 + h], aa = a1[gb * H1 + h];
        float sp = bf2f(s1_sh[m * H1 + h]);
        float vn = al * vv + (1.f - al) * I1 - sp - aa;   // THRESH = 1
        float sn = (vn - 1.f >= 0.f) ? 1.f : 0.f;
        v1[gb * H1 + h] = vn;
        a1[gb * H1 + h] = rh * aa + be * sn;
        s1_sh[m * H1 + h] = (sn > 0.f) ? (unsigned short)0x3F80 : (unsigned short)0;
      }
    }
    __syncthreads();

    // ---- layer 2: I2 = s1n @ W2.T + s2 @ W_rec2.T   (2 N-tiles per wave)
    v8f acc2[2];
    for (int tt = 0; tt < 2; ++tt) {
      const int n0 = (wave * 2 + tt) * 16;
      v8f c = V8ZERO;
      for (int k0 = 0; k0 < H1; k0 += 32) {
        v16bf A  = load_ab(&s1_sh[m16 * H1 + k0 + koff]);
        v16bf Bv = load_ab(&W2b[(n0 + m16) * H1 + k0 + koff]);
        c = WMMA_BF16(A, Bv, c);
      }
      for (int k0 = 0; k0 < H2; k0 += 32) {
        v16bf A  = load_ab(&s2_sh[m16 * H2 + k0 + koff]);
        v16bf Bv = load_ab(&Wr2[(n0 + m16) * H2 + k0 + koff]);
        c = WMMA_BF16(A, Bv, c);
      }
      acc2[tt] = c;
    }
    __syncthreads();

    // ---- layer 2 adLIF update + spike
    for (int tt = 0; tt < 2; ++tt) {
      const int n0 = (wave * 2 + tt) * 16;
      const int h  = n0 + m16;
      const float al = alpha2[h], rh = rho2[h], be = beta_a2[h];
      v8f c = acc2[tt];
      for (int r = 0; r < 8; ++r) {
        const int m = r + moff, gb = b0 + m;
        float I2 = c[r];
        float vv = v2[gb * H2 + h], aa = a2[gb * H2 + h];
        float sp = bf2f(s2_sh[m * H2 + h]);
        float vn = al * vv + (1.f - al) * I2 - sp - aa;
        float sn = (vn - 1.f >= 0.f) ? 1.f : 0.f;
        v2[gb * H2 + h] = vn;
        a2[gb * H2 + h] = rh * aa + be * sn;
        s2_sh[m * H2 + h] = (sn > 0.f) ? (unsigned short)0x3F80 : (unsigned short)0;
      }
    }
    __syncthreads();

    // ---- leaky readout (16x20 outputs, scalar dot over H2)
    if (tid < 16 * Osz) {
      int m = tid / Osz, o = tid % Osz;
      float io = 0.f;
      for (int h = 0; h < H2; h += 4) {
        io += bf2f(s2_sh[m * H2 + h + 0]) * Wout[o * H2 + h + 0]
            + bf2f(s2_sh[m * H2 + h + 1]) * Wout[o * H2 + h + 1]
            + bf2f(s2_sh[m * H2 + h + 2]) * Wout[o * H2 + h + 2]
            + bf2f(s2_sh[m * H2 + h + 3]) * Wout[o * H2 + h + 3];
      }
      float bo = beta_out[o];
      float vo = bo * vo_sh[tid] + (1.f - bo) * io;
      vo_sh[tid] = vo;
      acc_sh[tid] += vo;
    }
    __syncthreads();
  }

  if (tid < 16 * Osz) {
    int m = tid / Osz, o = tid % Osz;
    out[(b0 + m) * Osz + o] = acc_sh[tid] * (1.0f / (float)Tlen);
  }
}

// ---------------- host launcher ---------------------------------------------
extern "C" void kernel_launch(void* const* d_in, const int* in_sizes, int n_in,
                              void* d_out, int out_size, void* d_ws, size_t ws_size,
                              hipStream_t stream) {
  const float* x       = (const float*)d_in[0];
  const int*   delays  = (const int*)d_in[1];
  const float* W_in    = (const float*)d_in[2];
  const float* W_rec1  = (const float*)d_in[3];
  const float* W2      = (const float*)d_in[4];
  const float* W_rec2  = (const float*)d_in[5];
  const float* W_out   = (const float*)d_in[6];
  const float* alpha1  = (const float*)d_in[7];
  const float* rho1    = (const float*)d_in[8];
  const float* beta_a1 = (const float*)d_in[9];
  const float* alpha2  = (const float*)d_in[10];
  const float* rho2    = (const float*)d_in[11];
  const float* beta_a2 = (const float*)d_in[12];
  const float* beta_o  = (const float*)d_in[13];
  float* out = (float*)d_out;
  char*  ws  = (char*)d_ws;

  // workspace layout (bytes, all 16B aligned)
  constexpr size_t oWin = 0;
  constexpr size_t oWr1 = oWin + (size_t)H1 * Cpad * 2;       // 1.44 MB
  constexpr size_t oW2  = oWr1 + (size_t)H1 * H1 * 2;         // +2 MB
  constexpr size_t oWr2 = oW2  + (size_t)H2 * H1 * 2;         // +1 MB
  constexpr size_t oXd  = oWr2 + (size_t)H2 * H2 * 2;         // +0.5 MB
  constexpr size_t oIin = oXd  + (size_t)Bsz * Tpad * Cpad * 2;  // +92 MB
  constexpr size_t oV1  = oIin + (size_t)Bsz * Tpad * H1 * 4;    // +268 MB
  constexpr size_t oA1  = oV1  + (size_t)Bsz * H1 * 4;
  constexpr size_t oV2  = oA1  + (size_t)Bsz * H1 * 4;
  constexpr size_t oA2  = oV2  + (size_t)Bsz * H2 * 4;

  unsigned short* Winb = (unsigned short*)(ws + oWin);
  unsigned short* Wr1b = (unsigned short*)(ws + oWr1);
  unsigned short* W2b  = (unsigned short*)(ws + oW2);
  unsigned short* Wr2b = (unsigned short*)(ws + oWr2);
  unsigned short* xdb  = (unsigned short*)(ws + oXd);
  float* Iin = (float*)(ws + oIin);
  float* v1  = (float*)(ws + oV1);
  float* a1  = (float*)(ws + oA1);
  float* v2  = (float*)(ws + oV2);
  float* a2  = (float*)(ws + oA2);

  // 1) weights -> bf16
  k_convert_win<<<(H1 * Cpad + 255) / 256, 256, 0, stream>>>(W_in, Winb);
  k_f32_to_bf16<<<(H1 * H1 + 255) / 256, 256, 0, stream>>>(W_rec1, Wr1b, H1 * H1);
  k_f32_to_bf16<<<(H2 * H1 + 255) / 256, 256, 0, stream>>>(W2, W2b, H2 * H1);
  k_f32_to_bf16<<<(H2 * H2 + 255) / 256, 256, 0, stream>>>(W_rec2, Wr2b, H2 * H2);

  // 2) delayed input (binary -> exact bf16)
  long nxd = (long)Bsz * Tpad * Cpad;
  k_delay<<<(int)((nxd + 255) / 256), 256, 0, stream>>>(x, delays, xdb);

  // 3) input GEMM (fully parallel, WMMA)
  dim3 g(Bsz, Tpad / 16);
  k_input_gemm<<<g, 256, 0, stream>>>(xdb, Winb, Iin);

  // 4) persistent batch-partitioned recurrent scan (WMMA per step)
  k_recurrent<<<Bsz / 16, 512, 0, stream>>>(Iin, Wr1b, W2b, Wr2b, W_out,
                                            alpha1, rho1, beta_a1,
                                            alpha2, rho2, beta_a2, beta_o,
                                            v1, a1, v2, a2, out);
}
